// LPDecoder_43989055045968
// MI455X (gfx1250) — matrix operand order repaired and checked by reading, not compiled, not run on hardware
//
#include <hip/hip_runtime.h>
#include <hip/hip_bf16.h>
#include <hip/hip_fp16.h>

typedef _Float16 v4h  __attribute__((ext_vector_type(4)));
typedef _Float16 v8h  __attribute__((ext_vector_type(8)));
typedef _Float16 v16h __attribute__((ext_vector_type(16)));
typedef float    v8f  __attribute__((ext_vector_type(8)));

constexpr int IN_DIM = 128;   // per-node feature dim
constexpr int KDIM   = 256;   // concat(src,dst)
constexpr int HID    = 128;   // hidden dim
constexpr int STR    = 264;   // padded f16 stride (multiple of 8 -> 16B-aligned b128 chunks)
constexpr int WPB    = 8;     // waves per block (wave32)
constexpr int TPB    = WPB * 32;

// One wave computes sigmoid(relu([z_src||z_dst] @ W1 + b1) @ W2 + b2) for 16 edges
// using v_wmma_f32_16x16x32_f16: M=16 edges, N=128 hidden (8 tiles), K=256 (8 steps).
__global__ __launch_bounds__(TPB)
void lp_decoder_wmma(const float* __restrict__ z,
                     const int*   __restrict__ eidx,
                     const float* __restrict__ W1,
                     const float* __restrict__ b1,
                     const float* __restrict__ W2,
                     const float* __restrict__ b2,
                     float* __restrict__ out,
                     int E)
{
    // W1 transposed to [n][k] as f16 so a B-fragment is 2 contiguous ds_load_b128
    __shared__ alignas(16) _Float16 w1t[HID * STR];
    __shared__ float b1s[HID];
    __shared__ float w2s[HID];
    // per-wave A staging: 16 edges x 256 f16
    __shared__ alignas(16) _Float16 astg[WPB * 16 * STR];

    const int tid = threadIdx.x;

    // Stage W1 (global [k][n] row-major f32) -> LDS transposed f16
    for (int i = tid; i < KDIM * HID; i += TPB) {
        const int k = i >> 7;            // / HID
        const int n = i & (HID - 1);
        w1t[n * STR + k] = (_Float16)W1[i];
    }
    if (tid < HID) {
        b1s[tid] = b1[tid];
        w2s[tid] = W2[tid];
    }
    __syncthreads();

    const int lane = tid & 31;
    const int wave = tid >> 5;
    const int l16  = lane & 15;
    const int half = lane >> 4;          // 0: lanes 0-15, 1: lanes 16-31

    _Float16* aw = &astg[wave * 16 * STR];
    const float bias2 = b2[0];

    const int numTiles = E >> 4;
    const int gWave    = blockIdx.x * WPB + wave;
    const int nWaves   = gridDim.x * WPB;
    const int* __restrict__ srcA = eidx;
    const int* __restrict__ dstA = eidx + E;

    for (int tile = gWave; tile < numTiles; tile += nWaves) {
        const int ebase = tile << 4;

        // ---- Gather: 32 rows (16 edges x {src,dst}), batched 8 rows at a time
        // so 8 global_load_b128 are in flight before converting/storing to LDS.
#pragma unroll
        for (int g = 0; g < 4; ++g) {
            int    nodes[8];
            float4 v[8];
#pragma unroll
            for (int j = 0; j < 8; ++j) {
                const int i     = g * 8 + j;
                const int el    = i >> 1;
                const int which = i & 1;
                nodes[j] = which ? dstA[ebase + el] : srcA[ebase + el];
            }
#pragma unroll
            for (int j = 0; j < 8; ++j)
                v[j] = ((const float4*)(z + (size_t)nodes[j] * IN_DIM))[lane];
#pragma unroll
            for (int j = 0; j < 8; ++j) {
                const int i     = g * 8 + j;
                const int el    = i >> 1;
                const int which = i & 1;
                v4h h4;
                h4.x = (_Float16)v[j].x; h4.y = (_Float16)v[j].y;
                h4.z = (_Float16)v[j].z; h4.w = (_Float16)v[j].w;
                *(v4h*)&aw[el * STR + which * IN_DIM + lane * 4] = h4;
            }
        }

        // ---- A fragments (ISA 16-bit A 16x32 layout):
        // lane holds row M=l16; lanes 0-15 K-chunks {0..7,16..23}+kb*32,
        // lanes 16-31 {8..15,24..31}+kb*32 -> two 16B ds_load_b128 each.
        v16h afr[8];
#pragma unroll
        for (int kb = 0; kb < 8; ++kb) {
            const _Float16* p = &aw[l16 * STR + kb * 32 + half * 8];
            const v8h lo = *(const v8h*)p;
            const v8h hi = *(const v8h*)(p + 16);
            afr[kb] = __builtin_shufflevector(lo, hi,
                        0,1,2,3,4,5,6,7,8,9,10,11,12,13,14,15);
        }

        float part[8];
#pragma unroll
        for (int r = 0; r < 8; ++r) part[r] = 0.0f;

        // ---- 8 N-tiles x 8 K-steps of WMMA; bias+ReLU+W2 fused into epilogue.
        // unroll 2 (not 8): two independent WMMA chains in flight, B fragments
        // re-read from LDS each tile so nothing gets hoisted into scratch spills.
#pragma unroll 2
        for (int t = 0; t < 8; ++t) {
            v8f acc = {};
            // B fragment (32x16): lane holds column n=t*16+l16;
            // lanes 0-15: K {0..7,16..23}, lanes 16-31: K {8..15,24..31} (+kb*32), matching A.
            const _Float16* wcol = &w1t[(t * 16 + l16) * STR + half * 8];
#pragma unroll
            for (int kb = 0; kb < 8; ++kb) {
                const _Float16* q = wcol + kb * 32;
                const v8h lo = *(const v8h*)q;
                const v8h hi = *(const v8h*)(q + 16);
                const v16h bfr = __builtin_shufflevector(lo, hi,
                        0,1,2,3,4,5,6,7,8,9,10,11,12,13,14,15);
                acc = __builtin_amdgcn_wmma_f32_16x16x32_f16(
                        false, afr[kb], false, bfr, (short)0, acc, false, false);
            }
            const float b1v = b1s[t * 16 + l16];
            const float w2v = w2s[t * 16 + l16];
            // C/D layout: VGPR r, lane L -> h[M=r+8*half][n=t*16+l16]
#pragma unroll
            for (int r = 0; r < 8; ++r) {
                float h = acc[r] + b1v;
                h = fmaxf(h, 0.0f);
                part[r] = fmaf(h, w2v, part[r]);
            }
        }

        // ---- reduce columns across each 16-lane half (butterfly), sigmoid, store
#pragma unroll
        for (int r = 0; r < 8; ++r) {
            float s = part[r];
            s += __shfl_xor(s, 1, 32);
            s += __shfl_xor(s, 2, 32);
            s += __shfl_xor(s, 4, 32);
            s += __shfl_xor(s, 8, 32);
            part[r] = s;
        }
        if (l16 == 0) {
            const int rowBase = ebase + half * 8;   // lane0 rows 0-7, lane16 rows 8-15
#pragma unroll
            for (int r = 0; r < 8; ++r) {
                const float x = part[r] + bias2;
                out[rowBase + r] = 1.0f / (1.0f + __expf(-x));
            }
        }
    }
}

// Scalar fallback for E % 16 tail edges (E = 1M -> unused, but kept for generality)
__global__ void lp_decoder_tail(const float* __restrict__ z,
                                const int*   __restrict__ eidx,
                                const float* __restrict__ W1,
                                const float* __restrict__ b1,
                                const float* __restrict__ W2,
                                const float* __restrict__ b2,
                                float* __restrict__ out,
                                int E, int start)
{
    const int e = start + blockIdx.x * blockDim.x + threadIdx.x;
    if (e >= E) return;
    const float* zs = z + (size_t)eidx[e] * IN_DIM;
    const float* zd = z + (size_t)eidx[E + e] * IN_DIM;
    float s = b2[0];
    for (int n = 0; n < HID; ++n) {
        float a = b1[n];
        for (int k = 0; k < IN_DIM; ++k) a = fmaf(zs[k], W1[k * HID + n], a);
        for (int k = 0; k < IN_DIM; ++k) a = fmaf(zd[k], W1[(IN_DIM + k) * HID + n], a);
        a = fmaxf(a, 0.0f);
        s = fmaf(a, W2[n], s);
    }
    out[e] = 1.0f / (1.0f + __expf(-s));
}

extern "C" void kernel_launch(void* const* d_in, const int* in_sizes, int n_in,
                              void* d_out, int out_size, void* d_ws, size_t ws_size,
                              hipStream_t stream) {
    const float* z   = (const float*)d_in[0];
    const int*   eix = (const int*)  d_in[1];
    const float* W1  = (const float*)d_in[2];
    const float* b1  = (const float*)d_in[3];
    const float* W2  = (const float*)d_in[4];
    const float* b2  = (const float*)d_in[5];
    float* out = (float*)d_out;

    const int E        = in_sizes[1] / 2;
    const int numTiles = E >> 4;

    if (numTiles > 0) {
        int blocks = (numTiles + WPB - 1) / WPB;
        if (blocks > 1250) blocks = 1250;   // ~10k waves; grid-stride over tiles
        lp_decoder_wmma<<<blocks, TPB, 0, stream>>>(z, eix, W1, b1, W2, b2, out, E);
    }
    const int start = numTiles << 4;
    const int rem   = E - start;
    if (rem > 0) {
        lp_decoder_tail<<<(rem + 63) / 64, 64, 0, stream>>>(z, eix, W1, b1, W2, b2,
                                                            out, E, start);
    }
}
